// GNN1_72645076844714
// MI455X (gfx1250) — matrix operand order repaired and checked by reading, not compiled, not run on hardware
//
#include <hip/hip_runtime.h>
#include <hip/hip_bf16.h>

// ---------------------------------------------------------------------------
// CDNA5 (gfx1250) GNN pipeline, round 2.
// Dense contractions: v_wmma_f32_16x16x32_bf16, wave32, f32 accumulate.
//  - no per-element K guards (K multiple of 32; edge GEMM K=16 zero-padded)
//  - A rows clamped (loads always legal), C store guarded
//  - A fragments loaded as float4 runs; B pre-packed to bf16 lane layout
//  - 4 N-tiles per wave to reuse the A fragment (4 WMMAs / K-step)
// Scatter/softmax stages stay VALU (memory/atomic bound).
// ---------------------------------------------------------------------------

typedef __attribute__((ext_vector_type(16))) __bf16 v16bf;
typedef __attribute__((ext_vector_type(8)))  float  v8f;

__device__ __forceinline__ float gelu_t(float x) {
    const float k0 = 0.7978845608028654f, k1 = 0.044715f;
    float x3 = x * x * x;
    return 0.5f * x * (1.0f + tanhf(k0 * (x + k1 * x3)));
}

// Load one A fragment (16x32 bf16, ISA 7.12.2 layout) from f32 row-major.
// Row must be valid; K offset k0 must be a multiple of 32; row 16B-aligned.
__device__ __forceinline__ v16bf load_a_frag(const float* __restrict__ Arow,
                                             int k0, int h) {
    const float* p = Arow + k0 + h * 8;
    float4 a0 = *(const float4*)(p);
    float4 a1 = *(const float4*)(p + 4);
    float4 a2 = *(const float4*)(p + 16);
    float4 a3 = *(const float4*)(p + 20);
    v16bf a;
    a[0]  = (__bf16)a0.x; a[1]  = (__bf16)a0.y; a[2]  = (__bf16)a0.z; a[3]  = (__bf16)a0.w;
    a[4]  = (__bf16)a1.x; a[5]  = (__bf16)a1.y; a[6]  = (__bf16)a1.z; a[7]  = (__bf16)a1.w;
    a[8]  = (__bf16)a2.x; a[9]  = (__bf16)a2.y; a[10] = (__bf16)a2.z; a[11] = (__bf16)a2.w;
    a[12] = (__bf16)a3.x; a[13] = (__bf16)a3.y; a[14] = (__bf16)a3.z; a[15] = (__bf16)a3.w;
    return a;
}

// ---------------------------------------------------------------------------
// Pack f32 row-major B[K,N] into bf16 WMMA-B lane fragments:
// fragment index = (kt * (N/16) + nt) * 32 + lane, 16 bf16 (32 B) each.
// K rows beyond K are zero-padded (per 32-row K-tile).
// ---------------------------------------------------------------------------
__global__ void pack_b_kernel(const float* __restrict__ B,
                              __bf16* __restrict__ Bp, int K, int N) {
    int ktc = (K + 31) / 32;
    int total = ktc * (N >> 4) * 32;
    int t = blockIdx.x * blockDim.x + threadIdx.x;
    if (t >= total) return;
    int lane = t & 31;
    int rest = t >> 5;
    int nt = rest % (N >> 4);
    int kt = rest / (N >> 4);
    int h = lane >> 4, r = lane & 15;
    int col = nt * 16 + r;
    v16bf frag;
#pragma unroll
    for (int i = 0; i < 16; ++i) {
        int k = kt * 32 + ((i < 8) ? h * 8 + i : 16 + h * 8 + (i - 8));
        frag[i] = (__bf16)((k < K) ? B[(size_t)k * N + col] : 0.0f);
    }
    *(v16bf*)(Bp + (size_t)t * 16) = frag;
}

// ---------------------------------------------------------------------------
// GEMM: C[M,N] = act(A[M,K] @ B[K,N] + bias). One wave -> 16x64 output
// (4 n-tiles sharing one A fragment). Requires K%32==0, N%64==0.
// ---------------------------------------------------------------------------
__global__ void gemm_bias_kernel(const float* __restrict__ A,
                                 const __bf16* __restrict__ Bp,
                                 const float* __restrict__ bias,
                                 float* __restrict__ C,
                                 int M, int K, int N, int act) {
    int ngroups = N >> 6;
    int wave = blockIdx.x * (blockDim.x >> 5) + (threadIdx.x >> 5);
    int mt = wave / ngroups;
    int gidx = wave - mt * ngroups;
    if (mt * 16 >= M) return;

    int lane = threadIdx.x & 31;
    int h = lane >> 4, r = lane & 15;
    int ntiles = N >> 4;

    int mRow = mt * 16 + r;
    if (mRow >= M) mRow = M - 1;            // clamp: safe, store is guarded
    const float* Arow = A + (size_t)mRow * K;
    const v16bf* Bv = (const v16bf*)Bp;

    v8f acc0 = {}, acc1 = {}, acc2 = {}, acc3 = {};
    int kt = 0;
    for (int k0 = 0; k0 < K; k0 += 32, ++kt) {
        v16bf a = load_a_frag(Arow, k0, h);
        size_t base = ((size_t)kt * ntiles + gidx * 4) * 32 + lane;
        v16bf b0 = Bv[base];
        v16bf b1 = Bv[base + 32];
        v16bf b2 = Bv[base + 64];
        v16bf b3 = Bv[base + 96];
        acc0 = __builtin_amdgcn_wmma_f32_16x16x32_bf16(false, a, false, b0, (short)0, acc0, false, false);
        acc1 = __builtin_amdgcn_wmma_f32_16x16x32_bf16(false, a, false, b1, (short)0, acc1, false, false);
        acc2 = __builtin_amdgcn_wmma_f32_16x16x32_bf16(false, a, false, b2, (short)0, acc2, false, false);
        acc3 = __builtin_amdgcn_wmma_f32_16x16x32_bf16(false, a, false, b3, (short)0, acc3, false, false);
    }

#pragma unroll
    for (int j = 0; j < 4; ++j) {
        v8f acc = (j == 0) ? acc0 : (j == 1) ? acc1 : (j == 2) ? acc2 : acc3;
        int col = (gidx * 4 + j) * 16 + r;
        float bv = bias ? bias[col] : 0.0f;
#pragma unroll
        for (int rr = 0; rr < 8; ++rr) {
            int m = mt * 16 + h * 8 + rr;
            if (m < M) {
                float v = acc[rr] + bv;
                if (act == 1) v = gelu_t(v);
                C[(size_t)m * N + col] = v;
            }
        }
    }
}

// ---------------------------------------------------------------------------
// Fused edge-message kernel: tile of 16 edges,
//   msg = edge_attr[e](K=16, padded to 32) @ W_edge1 + b_edge1 + xm[src[e]]
//   atomicAdd into h[dst[e]].  W_edge1 pre-packed (ktc=1, ntiles=16).
// ---------------------------------------------------------------------------
__global__ void edge_msg_kernel(const float* __restrict__ edge_attr,
                                const __bf16* __restrict__ We1p,
                                const float* __restrict__ b_edge1,
                                const float* __restrict__ xm,
                                const int* __restrict__ src,
                                const int* __restrict__ dst,
                                float* __restrict__ hbuf, int E) {
    int wave = blockIdx.x * (blockDim.x >> 5) + (threadIdx.x >> 5);
    int e0 = wave * 16;
    if (e0 >= E) return;
    int lane = threadIdx.x & 31;
    int h = lane >> 4, r = lane & 15;

    // A fragment: row = edge e0+r (clamped), K = h*8..h*8+7 valid, rest zero
    int aRow = e0 + r;
    if (aRow >= E) aRow = E - 1;
    const float* p = edge_attr + (size_t)aRow * 16 + h * 8;
    float4 q0 = *(const float4*)(p);
    float4 q1 = *(const float4*)(p + 4);
    v16bf a;
    a[0] = (__bf16)q0.x; a[1] = (__bf16)q0.y; a[2] = (__bf16)q0.z; a[3] = (__bf16)q0.w;
    a[4] = (__bf16)q1.x; a[5] = (__bf16)q1.y; a[6] = (__bf16)q1.z; a[7] = (__bf16)q1.w;
#pragma unroll
    for (int t = 8; t < 16; ++t) a[t] = (__bf16)0.0f;

    // edge ids for this lane's 8 C rows
    int ev[8], sv[8], dv[8];
#pragma unroll
    for (int rr = 0; rr < 8; ++rr) {
        int e = e0 + h * 8 + rr;
        ev[rr] = e;
        int ec = (e < E) ? e : (E - 1);
        sv[rr] = src[ec];
        dv[rr] = dst[ec];
    }

    const v16bf* Bv = (const v16bf*)We1p;
    for (int nt = 0; nt < 16; ++nt) {
        v16bf b = Bv[(size_t)nt * 32 + lane];
        v8f acc = {};
        acc = __builtin_amdgcn_wmma_f32_16x16x32_bf16(false, a, false, b, (short)0, acc, false, false);
        int col = nt * 16 + r;
        float bias = b_edge1[col];
#pragma unroll
        for (int rr = 0; rr < 8; ++rr) {
            if (ev[rr] < E) {
                float val = acc[rr] + bias + xm[(size_t)sv[rr] * 256 + col];
                atomicAdd(&hbuf[(size_t)dv[rr] * 256 + col], val);
            }
        }
    }
}

// ---------------------------------------------------------------------------
__global__ void gelu_kernel(float* __restrict__ x, size_t n) {
    size_t i = blockIdx.x * (size_t)blockDim.x + threadIdx.x;
    size_t stride = (size_t)gridDim.x * blockDim.x;
    for (; i < n; i += stride) x[i] = gelu_t(x[i]);
}

__global__ void edge_aux_kernel(const int* __restrict__ dst,
                                const float* __restrict__ edge_attr,
                                float* __restrict__ loopsum,
                                float* __restrict__ deg, int E) {
    long long t = blockIdx.x * (long long)blockDim.x + threadIdx.x;
    if (t >= (long long)E * 16) return;
    int e = (int)(t >> 4), f = (int)(t & 15);
    int d = dst[e];
    atomicAdd(&loopsum[(size_t)d * 16 + f], edge_attr[(size_t)e * 16 + f]);
    if (f == 0) atomicAdd(&deg[d], 1.0f);
}

__global__ void loop_div_kernel(float* __restrict__ loopsum,
                                const float* __restrict__ deg, int Nn) {
    int t = blockIdx.x * blockDim.x + threadIdx.x;
    if (t >= Nn * 16) return;
    loopsum[t] /= fmaxf(deg[t >> 4], 1.0f);
}

// ---------------------------------------------------------------------------
// Attention pass A: alpha = att . leaky_relu(xl[s]+xr[d]+ea@W_e2), per-dst
// atomicMax via order-preserving uint encoding. One wave per edge.
// ---------------------------------------------------------------------------
__global__ void alpha_kernel(const int* __restrict__ src,
                             const int* __restrict__ dst,
                             const float* __restrict__ edge_attr,
                             const float* __restrict__ loop_attr,
                             const float* __restrict__ xl,
                             const float* __restrict__ xr,
                             const float* __restrict__ W_e2,
                             const float* __restrict__ att,
                             float* __restrict__ alpha,
                             unsigned* __restrict__ amaxu, int E, int Nn) {
    int idx = blockIdx.x * (blockDim.x >> 5) + (threadIdx.x >> 5);
    int T = E + Nn;
    if (idx >= T) return;
    int lane = threadIdx.x & 31;
    int s, d;
    const float* ea;
    if (idx < E) { s = src[idx]; d = dst[idx]; ea = edge_attr + (size_t)idx * 16; }
    else { s = idx - E; d = s; ea = loop_attr + (size_t)s * 16; }

    float eareg[16];
#pragma unroll
    for (int k = 0; k < 16; ++k) eareg[k] = ea[k];

    float partial = 0.0f;
#pragma unroll
    for (int j = 0; j < 4; ++j) {
        int c = lane + 32 * j;
        float v = xl[(size_t)s * 128 + c] + xr[(size_t)d * 128 + c];
#pragma unroll
        for (int k = 0; k < 16; ++k) v += eareg[k] * W_e2[k * 128 + c];
        v = (v > 0.0f) ? v : 0.2f * v;
        partial += v * att[c];
    }
    for (int off = 16; off > 0; off >>= 1) partial += __shfl_xor(partial, off, 32);
    if (lane == 0) {
        alpha[idx] = partial;
        unsigned u = __float_as_uint(partial);
        u = (u & 0x80000000u) ? ~u : (u | 0x80000000u);
        atomicMax(&amaxu[d], u);
    }
}

__global__ void expsum_kernel(const int* __restrict__ dst,
                              float* __restrict__ alpha,
                              const unsigned* __restrict__ amaxu,
                              float* __restrict__ denom, int E, int Nn) {
    int idx = blockIdx.x * blockDim.x + threadIdx.x;
    int T = E + Nn;
    if (idx >= T) return;
    int d = (idx < E) ? dst[idx] : (idx - E);
    unsigned u = amaxu[d];
    float amax = (u & 0x80000000u) ? __uint_as_float(u ^ 0x80000000u)
                                   : __uint_as_float(~u);
    float e = __expf(alpha[idx] - amax);
    alpha[idx] = e;
    atomicAdd(&denom[d], e);
}

__global__ void scatter_out_kernel(const int* __restrict__ src,
                                   const int* __restrict__ dst,
                                   const float* __restrict__ alpha,
                                   const float* __restrict__ denom,
                                   const float* __restrict__ xl,
                                   float* __restrict__ outm, int E, int Nn) {
    int idx = blockIdx.x * (blockDim.x >> 5) + (threadIdx.x >> 5);
    int T = E + Nn;
    if (idx >= T) return;
    int lane = threadIdx.x & 31;
    int s, d;
    if (idx < E) { s = src[idx]; d = dst[idx]; } else { s = idx - E; d = s; }
    float a = alpha[idx] / denom[d];
#pragma unroll
    for (int j = 0; j < 4; ++j) {
        int c = lane + 32 * j;
        atomicAdd(&outm[(size_t)d * 128 + c], xl[(size_t)s * 128 + c] * a);
    }
}

// ---------------------------------------------------------------------------
__global__ void pool_kernel(const int* __restrict__ batch,
                            const float* __restrict__ outm,
                            float* __restrict__ gsum,
                            float* __restrict__ cnt, int Nn) {
    int i = blockIdx.x * (blockDim.x >> 5) + (threadIdx.x >> 5);
    if (i >= Nn) return;
    int lane = threadIdx.x & 31;
    int b = batch[i];
#pragma unroll
    for (int j = 0; j < 4; ++j) {
        int c = lane + 32 * j;
        atomicAdd(&gsum[(size_t)b * 128 + c], outm[(size_t)i * 128 + c]);
    }
    if (lane == 0) atomicAdd(&cnt[b], 1.0f);
}

__global__ void pool_div_kernel(const float* __restrict__ gsum,
                                const float* __restrict__ cnt,
                                float* __restrict__ g, int G) {
    int t = blockIdx.x * blockDim.x + threadIdx.x;
    if (t >= G * 128) return;
    g[t] = gsum[t] / fmaxf(cnt[t >> 7], 1.0f);
}

// Final y[g] = hidden[g] @ W2 + b2. One wave per graph.
__global__ void out_gemv_kernel(const float* __restrict__ hidden,
                                const float* __restrict__ W2,
                                const float* __restrict__ b2,
                                float* __restrict__ y, int G) {
    int g = blockIdx.x * (blockDim.x >> 5) + (threadIdx.x >> 5);
    if (g >= G) return;
    int lane = threadIdx.x & 31;
    float p = 0.0f;
#pragma unroll
    for (int j = 0; j < 16; ++j) {
        int c = lane + 32 * j;
        p += hidden[(size_t)g * 512 + c] * W2[c];
    }
    for (int off = 16; off > 0; off >>= 1) p += __shfl_xor(p, off, 32);
    if (lane == 0) y[g] = p + b2[0];
}

// ---------------------------------------------------------------------------
__global__ void init_kernel(float* deg, float* loopsum, float* denom,
                            unsigned* amaxu, float* outm,
                            const float* __restrict__ bias2,
                            float* gsum, float* cnt, int Nn, int G) {
    size_t tid = blockIdx.x * (size_t)blockDim.x + threadIdx.x;
    size_t stride = (size_t)gridDim.x * blockDim.x;
    for (size_t i = tid; i < (size_t)Nn; i += stride) {
        deg[i] = 0.0f; denom[i] = 0.0f; amaxu[i] = 0u;
    }
    for (size_t i = tid; i < (size_t)Nn * 16; i += stride) loopsum[i] = 0.0f;
    for (size_t i = tid; i < (size_t)Nn * 128; i += stride) outm[i] = bias2[i & 127];
    for (size_t i = tid; i < (size_t)G * 128; i += stride) gsum[i] = 0.0f;
    for (size_t i = tid; i < (size_t)G; i += stride) cnt[i] = 0.0f;
}

// ---------------------------------------------------------------------------
extern "C" void kernel_launch(void* const* d_in, const int* in_sizes, int n_in,
                              void* d_out, int out_size, void* d_ws, size_t ws_size,
                              hipStream_t stream) {
    const float* x         = (const float*)d_in[0];
    const int*   edge_idx  = (const int*)d_in[1];   // [2,E]: row0 src, row1 dst
    const float* edge_attr = (const float*)d_in[2];
    const int*   batch     = (const int*)d_in[3];
    const float* W_msg   = (const float*)d_in[4];
    const float* b_msg   = (const float*)d_in[5];
    const float* W_edge1 = (const float*)d_in[6];
    const float* b_edge1 = (const float*)d_in[7];
    const float* W_self  = (const float*)d_in[8];
    const float* b_self  = (const float*)d_in[9];
    const float* W_l   = (const float*)d_in[10];
    const float* b_l   = (const float*)d_in[11];
    const float* W_r   = (const float*)d_in[12];
    const float* b_r   = (const float*)d_in[13];
    const float* W_e2  = (const float*)d_in[14];
    const float* att   = (const float*)d_in[15];
    const float* bias2 = (const float*)d_in[16];
    const float* W1 = (const float*)d_in[17];
    const float* b1 = (const float*)d_in[18];
    const float* W2 = (const float*)d_in[19];
    const float* b2 = (const float*)d_in[20];
    float* y = (float*)d_out;

    const int Nn = in_sizes[0] / 64;    // 40000
    const int E  = in_sizes[2] / 16;    // 640000
    const int G  = out_size;            // 1000
    const int T  = E + Nn;

    const int* src = edge_idx;
    const int* dst = edge_idx + E;

    char* p = (char*)d_ws;
    auto carve = [&](size_t bytes) -> char* {
        char* r = p;
        p += (bytes + 255) & ~(size_t)255;
        return r;
    };
    auto packed_elems = [](int K, int N) -> size_t {
        return (size_t)((K + 31) / 32) * (N / 16) * 32 * 16;
    };

    float*    xm      = (float*)carve((size_t)Nn * 256 * 4);
    float*    hbuf    = (float*)carve((size_t)Nn * 256 * 4);
    float*    xl      = (float*)carve((size_t)Nn * 128 * 4);
    float*    xr      = (float*)carve((size_t)Nn * 128 * 4);
    float*    outm    = (float*)carve((size_t)Nn * 128 * 4);
    float*    loopsum = (float*)carve((size_t)Nn * 16 * 4);
    float*    deg     = (float*)carve((size_t)Nn * 4);
    float*    denom   = (float*)carve((size_t)Nn * 4);
    unsigned* amaxu   = (unsigned*)carve((size_t)Nn * 4);
    float*    alphab  = (float*)carve((size_t)T * 4);
    float*    gsum    = (float*)carve((size_t)G * 128 * 4);
    float*    cnt     = (float*)carve((size_t)G * 4);
    float*    gbuf    = (float*)carve((size_t)G * 128 * 4);
    float*    hidden  = (float*)carve((size_t)G * 512 * 4);
    __bf16* Wmsgp  = (__bf16*)carve(packed_elems(64, 256) * 2);
    __bf16* Wselfp = (__bf16*)carve(packed_elems(64, 256) * 2);
    __bf16* We1p   = (__bf16*)carve(packed_elems(16, 256) * 2);
    __bf16* Wlp    = (__bf16*)carve(packed_elems(256, 128) * 2);
    __bf16* Wrp    = (__bf16*)carve(packed_elems(256, 128) * 2);
    __bf16* W1p    = (__bf16*)carve(packed_elems(128, 512) * 2);
    (void)ws_size; (void)n_in;

    const int TB = 256;  // 8 wave32 waves per block
    auto gemm_blocks = [&](int M, int N) -> int {
        long long waves = ((long long)(M + 15) / 16) * (N / 64);
        return (int)((waves * 32 + TB - 1) / TB);
    };
    auto pack_blocks = [&](int K, int N) -> int {
        int total = ((K + 31) / 32) * (N / 16) * 32;
        return (total + TB - 1) / TB;
    };

    // 0. init scratch + pack weights to bf16 WMMA-B layout
    init_kernel<<<4096, TB, 0, stream>>>(deg, loopsum, denom, amaxu, outm,
                                         bias2, gsum, cnt, Nn, G);
    pack_b_kernel<<<pack_blocks(64, 256), TB, 0, stream>>>(W_msg,   Wmsgp,  64, 256);
    pack_b_kernel<<<pack_blocks(64, 256), TB, 0, stream>>>(W_self,  Wselfp, 64, 256);
    pack_b_kernel<<<pack_blocks(16, 256), TB, 0, stream>>>(W_edge1, We1p,   16, 256);
    pack_b_kernel<<<pack_blocks(256,128), TB, 0, stream>>>(W_l,     Wlp,   256, 128);
    pack_b_kernel<<<pack_blocks(256,128), TB, 0, stream>>>(W_r,     Wrp,   256, 128);
    pack_b_kernel<<<pack_blocks(128,512), TB, 0, stream>>>(W1,      W1p,   128, 512);

    // 1. xm = x @ W_msg + b_msg ; h = x @ W_self + b_self   (WMMA)
    gemm_bias_kernel<<<gemm_blocks(Nn, 256), TB, 0, stream>>>(
        x, Wmsgp, b_msg, xm, Nn, 64, 256, 0);
    gemm_bias_kernel<<<gemm_blocks(Nn, 256), TB, 0, stream>>>(
        x, Wselfp, b_self, hbuf, Nn, 64, 256, 0);

    // 2. fused edge message GEMM + gather(xm[src]) + scatter-add h[dst] (WMMA)
    {
        int tiles = (E + 15) / 16;
        edge_msg_kernel<<<(tiles + 7) / 8, TB, 0, stream>>>(
            edge_attr, We1p, b_edge1, xm, src, dst, hbuf, E);
    }

    // 3. h = gelu(h)
    gelu_kernel<<<4096, TB, 0, stream>>>(hbuf, (size_t)Nn * 256);

    // 4. deg + loop_attr (self-loop edge_attr = mean of incoming)
    {
        long long t = (long long)E * 16;
        edge_aux_kernel<<<(int)((t + TB - 1) / TB), TB, 0, stream>>>(
            dst, edge_attr, loopsum, deg, E);
        loop_div_kernel<<<(Nn * 16 + TB - 1) / TB, TB, 0, stream>>>(loopsum, deg, Nn);
    }

    // 5. xl = h @ W_l + b_l ; xr = h @ W_r + b_r   (WMMA)
    gemm_bias_kernel<<<gemm_blocks(Nn, 128), TB, 0, stream>>>(
        hbuf, Wlp, b_l, xl, Nn, 256, 128, 0);
    gemm_bias_kernel<<<gemm_blocks(Nn, 128), TB, 0, stream>>>(
        hbuf, Wrp, b_r, xr, Nn, 256, 128, 0);

    // 6. attention: alpha -> segment max -> exp/sum -> weighted scatter
    alpha_kernel<<<(T + 7) / 8, TB, 0, stream>>>(
        src, dst, edge_attr, loopsum, xl, xr, W_e2, att, alphab, amaxu, E, Nn);
    expsum_kernel<<<(T + TB - 1) / TB, TB, 0, stream>>>(
        dst, alphab, amaxu, denom, E, Nn);
    scatter_out_kernel<<<(T + 7) / 8, TB, 0, stream>>>(
        src, dst, alphab, denom, xl, outm, E, Nn);

    // 7. global mean pool -> g
    pool_kernel<<<(Nn + 7) / 8, TB, 0, stream>>>(batch, outm, gsum, cnt, Nn);
    pool_div_kernel<<<(G * 128 + TB - 1) / TB, TB, 0, stream>>>(gsum, cnt, gbuf, G);

    // 8. FFN: hidden = gelu(g @ W1 + b1)  (WMMA) ; y = hidden @ W2 + b2
    gemm_bias_kernel<<<gemm_blocks(G, 512), TB, 0, stream>>>(
        gbuf, W1p, b1, hidden, G, 128, 512, 1);
    out_gemv_kernel<<<(G + 7) / 8, TB, 0, stream>>>(hidden, W2, b2, y, G);
}
